// SpResNetD4HD_19937238188233
// MI455X (gfx1250) — matrix operand order repaired and checked by reading, not compiled, not run on hardware
//
#include <hip/hip_runtime.h>
#include <hip/hip_bf16.h>

typedef __attribute__((ext_vector_type(16))) __bf16        v16bf;
typedef __attribute__((ext_vector_type(8)))  float         v8f;
typedef __attribute__((ext_vector_type(4)))  unsigned int  u32x4;

#define STATS_NB 1024

typedef __hip_bfloat16 bf16;

// ----------------------------------------------------------------------------
// Scatter voxel features into dense NDHWC bf16 grid + f32 mask
// ----------------------------------------------------------------------------
__global__ void k_scatter(const float* __restrict__ vf, const int* __restrict__ coors,
                          int N, int H, int W, int Cin,
                          bf16* __restrict__ grid, float* __restrict__ mask)
{
    int i = blockIdx.x * blockDim.x + threadIdx.x;
    if (i >= N) return;
    int z = coors[4*i + 1], y = coors[4*i + 2], x = coors[4*i + 3];
    long v = ((long)(z * H + y) * W + x);
    mask[v] = 1.0f;
    for (int c = 0; c < Cin; ++c)
        grid[v * Cin + c] = __float2bfloat16(vf[(long)i * Cin + c]);
}

// ----------------------------------------------------------------------------
// Convert f32 weights [KD,KH,KW,Cin,Cout] -> bf16 [tap][Cout][Cin]
// ----------------------------------------------------------------------------
__global__ void k_wconvert(const float* __restrict__ w, int TAP, int Cin, int Cout,
                           bf16* __restrict__ wT)
{
    int i = blockIdx.x * blockDim.x + threadIdx.x;
    int total = TAP * Cin * Cout;
    if (i >= total) return;
    int co  = i % Cout;
    int t   = i / Cout;
    int ci  = t % Cin;
    int tap = t / Cin;
    wT[((long)tap * Cout + co) * Cin + ci] = __float2bfloat16(w[i]);
}

// ----------------------------------------------------------------------------
// Implicit-GEMM 3D conv, bf16 in / f32 accumulate via v_wmma_f32_16x16x32_bf16.
// One wave computes a 16-voxel x COUT tile (NT = COUT/16 accumulators); the A
// fragment is loaded once per tap/K-chunk and reused for all NT WMMAs.
//   A (16x32, MxK): lanes 0-15 row M=l, dwords0-3=K[base..base+7], 4-7=K[base+8..base+15]
//                   base = 0 (lanes<16) / 16 (lanes>=16)
//   B (32x16, KxN): lane col N=l&15, 16 contiguous K, +16 for lanes>=16
//   C (16x16 f32):  VGPR r, lanes<16 -> M=r, lanes>=16 -> M=r+8, N=l&15
// Grid: (ceil(Wo/16), Ho, Do); blockDim 32.
// ----------------------------------------------------------------------------
template<int CIN, int COUT, int KD, int KH, int KW>
__global__ __launch_bounds__(32)
void k_conv_wmma(const bf16* __restrict__ x, const bf16* __restrict__ wT,
                 bf16* __restrict__ y,
                 int Di, int Hi, int Wi,
                 int Do, int Ho, int Wo,
                 int sd, int sh, int sw, int pd, int ph, int pw)
{
    constexpr int NT = COUT / 16;
    const int lane = threadIdx.x;
    const int oy = blockIdx.y;
    const int oz = blockIdx.z;
    const int x0 = blockIdx.x << 4;
    const int m  = lane & 15;          // A row / B column within tile
    const int hi = lane >> 4;          // half-wave select
    const int ox = x0 + m;

    const int abase1 = hi ? 8  : 0;    // unused: A also loads from bbase
    const int abase2 = 16 + abase1;    // unused
    const int bbase  = hi ? 16 : 0;    // B: K start for this half-wave

    v8f acc[NT];
#pragma unroll
    for (int t = 0; t < NT; ++t) acc[t] = (v8f){0.f,0.f,0.f,0.f,0.f,0.f,0.f,0.f};

    for (int kd = 0; kd < KD; ++kd) {
        int iz = oz * sd + kd - pd;
        if (iz < 0 || iz >= Di) continue;
        for (int kh = 0; kh < KH; ++kh) {
            int iy = oy * sh + kh - ph;
            if (iy < 0 || iy >= Hi) continue;
#pragma unroll
            for (int kw = 0; kw < KW; ++kw) {
                int ix = ox * sw + kw - pw;
                bool inb = (ix >= 0) && (ix < Wi);
                long vbase = ((long)(iz * Hi + iy) * Wi + ix) * CIN;
                int  tap   = (kd * KH + kh) * KW + kw;
                const bf16* wtap = wT + (long)tap * COUT * CIN;

#pragma unroll
                for (int ci0 = 0; ci0 < CIN; ci0 += 32) {
                    union { v16bf v; u32x4 q[2]; } a;
                    a.q[0] = (u32x4){0u,0u,0u,0u};
                    a.q[1] = (u32x4){0u,0u,0u,0u};
                    if (inb) {
                        if constexpr (CIN >= 32) {
                            a.q[0] = *(const u32x4*)(x + vbase + ci0 + bbase);
                            a.q[1] = *(const u32x4*)(x + vbase + ci0 + bbase + 8);
                        } else {
                            // CIN==16: only K<16 exists; lanes>=16 (K 16..31) stay zero
                            if (!hi) {
                                a.q[0] = *(const u32x4*)(x + vbase);
                                a.q[1] = *(const u32x4*)(x + vbase + 8);
                            }
                        }
                    }
#pragma unroll
                    for (int t = 0; t < NT; ++t) {
                        union { v16bf v; u32x4 q[2]; } b;
                        b.q[0] = (u32x4){0u,0u,0u,0u};
                        b.q[1] = (u32x4){0u,0u,0u,0u};
                        const int co = (t << 4) + m;
                        if constexpr (CIN >= 32) {
                            const bf16* wp = wtap + (long)co * CIN + ci0 + bbase;
                            b.q[0] = *(const u32x4*)(wp);
                            b.q[1] = *(const u32x4*)(wp + 8);
                        } else {
                            if (!hi) {   // bbase==16 is out of range for CIN==16
                                const bf16* wp = wtap + (long)co * CIN;
                                b.q[0] = *(const u32x4*)(wp);
                                b.q[1] = *(const u32x4*)(wp + 8);
                            }
                        }
                        acc[t] = __builtin_amdgcn_wmma_f32_16x16x32_bf16(
                                     false, a.v, false, b.v, (short)0, acc[t], false, false);
                    }
                }
            }
        }
    }

    // Epilogue: scatter C fragments to NDHWC bf16 output (raw conv result)
    const int n = lane & 15;
#pragma unroll
    for (int t = 0; t < NT; ++t) {
#pragma unroll
        for (int r = 0; r < 8; ++r) {
            int mm  = r + (hi << 3);
            int oxx = x0 + mm;
            if (oxx < Wo) {
                long oidx = ((long)(oz * Ho + oy) * Wo + oxx) * COUT + (t << 4) + n;
                y[oidx] = __float2bfloat16(acc[t][r]);
            }
        }
    }
}

// ----------------------------------------------------------------------------
// Downsample mask:  newmask[o] = any(mask over receptive field) ? 1 : 0
// ----------------------------------------------------------------------------
__global__ void k_mask_down(const float* __restrict__ mi, float* __restrict__ mo,
                            int Di, int Hi, int Wi, int Do, int Ho, int Wo,
                            int KD, int KH, int KW,
                            int sd, int sh, int sw, int pd, int ph, int pw)
{
    int o = blockIdx.x * blockDim.x + threadIdx.x;
    int total = Do * Ho * Wo;
    if (o >= total) return;
    int ow = o % Wo; int t = o / Wo; int oh = t % Ho; int od = t / Ho;
    float v = 0.f;
    for (int kd = 0; kd < KD; ++kd) {
        int iz = od * sd + kd - pd; if (iz < 0 || iz >= Di) continue;
        for (int kh = 0; kh < KH; ++kh) {
            int iy = oh * sh + kh - ph; if (iy < 0 || iy >= Hi) continue;
            for (int kw = 0; kw < KW; ++kw) {
                int ix = ow * sw + kw - pw; if (ix < 0 || ix >= Wi) continue;
                if (mi[((long)(iz * Hi + iy) * Wi + ix)] > 0.f) v = 1.f;
            }
        }
    }
    mo[o] = v;
}

// ----------------------------------------------------------------------------
// BN stats pass 1: deterministic per-block partial sums of (y*mask), (y*mask)^2
// C = 1<<cshift (power of two). part layout/block: [C sums][C sumsq][1 count]
// ----------------------------------------------------------------------------
__global__ __launch_bounds__(256)
void k_stats_partial(const bf16* __restrict__ y, const float* __restrict__ mask,
                     long nvox, int cshift, float* __restrict__ part)
{
    __shared__ float s1[256], s2[256], sc[256];
    const int C = 1 << cshift;
    int tid = threadIdx.x;
    int c = tid & (C - 1);                // constant per thread (stride % C == 0)
    float sum = 0.f, sq = 0.f, cnt = 0.f;
    long total  = nvox << cshift;
    long stride = (long)gridDim.x * 256;
    for (long e = (long)blockIdx.x * 256 + tid; e < total; e += stride) {
        long v  = e >> cshift;
        float mk = mask[v];
        float val = __bfloat162float(y[e]) * mk;
        sum += val;
        sq  += val * val;
        if (c == 0) cnt += mk;
    }
    s1[tid] = sum; s2[tid] = sq; sc[tid] = cnt;
    __syncthreads();
    for (int off = 128; off >= C; off >>= 1) {
        if (tid < off) { s1[tid] += s1[tid+off]; s2[tid] += s2[tid+off]; sc[tid] += sc[tid+off]; }
        __syncthreads();
    }
    int stridep = 2 * C + 1;
    if (tid < C) {
        part[blockIdx.x * stridep + tid]     = s1[tid];
        part[blockIdx.x * stridep + C + tid] = s2[tid];
    }
    if (tid == 0) part[blockIdx.x * stridep + 2*C] = sc[0];
}

// ----------------------------------------------------------------------------
// BN stats pass 2: fold partials (fixed order), emit per-channel scale/shift
// ----------------------------------------------------------------------------
__global__ void k_stats_finalize(const float* __restrict__ part, int NB, int C,
                                 const float* __restrict__ g, const float* __restrict__ b,
                                 float* __restrict__ ss, float eps)
{
    int c = threadIdx.x;            // blockDim == C
    int stridep = 2 * C + 1;
    float sum = 0.f, sq = 0.f;
    for (int i = 0; i < NB; ++i) {
        sum += part[i * stridep + c];
        sq  += part[i * stridep + C + c];
    }
    __shared__ float cntS;
    if (c == 0) {
        float cnt = 0.f;
        for (int i = 0; i < NB; ++i) cnt += part[i * stridep + 2*C];
        cntS = fmaxf(cnt, 1.0f);
    }
    __syncthreads();
    float cnt  = cntS;
    float mean = sum / cnt;
    float var  = sq / cnt - mean * mean;
    float sc   = g[c] * rsqrtf(var + eps);
    ss[c]      = sc;
    ss[C + c]  = b[c] - mean * sc;
}

// ----------------------------------------------------------------------------
// BN apply: z = relu?( (y*scale + shift)*mask [+ res] ), bf16 out
// ----------------------------------------------------------------------------
__global__ void k_bn_apply(const bf16* __restrict__ y, const float* __restrict__ mask,
                           const float* __restrict__ ss, int cshift, long nvox,
                           const bf16* __restrict__ res, int relu,
                           bf16* __restrict__ z)
{
    const int C = 1 << cshift;
    long e = (long)blockIdx.x * blockDim.x + threadIdx.x;
    if (e >= (nvox << cshift)) return;
    int  c = (int)(e & (C - 1));
    long v = e >> cshift;
    float mk  = mask[v];
    float val = (__bfloat162float(y[e]) * ss[c] + ss[C + c]) * mk;
    if (res)  val += __bfloat162float(res[e]);
    if (relu) val = fmaxf(val, 0.f);
    z[e] = __float2bfloat16(val);
}

// ----------------------------------------------------------------------------
// Final BN + relu + transpose (D,H,W,C) -> (C*D, H, W), f32 output
// ----------------------------------------------------------------------------
__global__ void k_bn_final(const bf16* __restrict__ y, const float* __restrict__ mask,
                           const float* __restrict__ ss, int cshift,
                           int Dd, int Hh, int Ww, float* __restrict__ out)
{
    const int C = 1 << cshift;
    int e = blockIdx.x * blockDim.x + threadIdx.x;
    int total = (Dd * Hh * Ww) << cshift;
    if (e >= total) return;
    int c = e & (C - 1); int v = e >> cshift;
    int w = v % Ww; int t = v / Ww; int h = t % Hh; int d = t / Hh;
    float mk  = mask[v];
    float val = (__bfloat162float(y[e]) * ss[c] + ss[C + c]) * mk;
    val = fmaxf(val, 0.f);
    out[((long)(c * Dd + d) * Hh + h) * Ww + w] = val;
}

// ============================================================================
extern "C" void kernel_launch(void* const* d_in, const int* in_sizes, int n_in,
                              void* d_out, int out_size, void* d_ws, size_t ws_size,
                              hipStream_t stream)
{
    // Resolutions
    const int D0=21, H0=400, W0=352;
    const int D1=11, H1=200, W1=176;
    const int D2=5,  H2=100, W2=88;
    const int D3=2,  H3=100, W3=88;
    const long V0=(long)D0*H0*W0, V1=(long)D1*H1*W1, V2=(long)D2*H2*W2, V3=(long)D3*H3*W3;

    struct CSpec { int KD,KH,KW,Cin,Cout; };
    static const CSpec SP[16] = {
        {3,3,3,16,32}, {3,3,3,32,32},{3,3,3,32,32},{3,3,3,32,32},{3,3,3,32,32},
        {3,3,3,32,64}, {3,3,3,64,64},{3,3,3,64,64},{3,3,3,64,64},{3,3,3,64,64},
        {3,3,3,64,64}, {3,3,3,64,64},{3,3,3,64,64},{3,3,3,64,64},{3,3,3,64,64},
        {3,1,1,64,64}
    };

    // ---- workspace carve-up (256B aligned) ----
    const size_t PSZ   = (size_t)V0 * 32;          // bf16 elements per ping buffer
    const size_t WSLOT = 27 * 64 * 64;             // bf16 elements per weight slot
    char* ws = (char*)d_ws;
    size_t off = 0;
    auto carve = [&](size_t bytes) -> char* {
        char* p = ws + off;
        off += (bytes + 255) & ~(size_t)255;
        return p;
    };
    bf16*  P[4];
    for (int i = 0; i < 4; ++i) P[i] = (bf16*)carve(PSZ * sizeof(bf16));
    float* mask0 = (float*)carve((size_t)V0 * 4);
    float* mask1 = (float*)carve((size_t)V1 * 4);
    float* mask2 = (float*)carve((size_t)V2 * 4);
    float* mask3 = (float*)carve((size_t)V3 * 4);
    bf16*  wbuf  = (bf16*) carve(WSLOT * 16 * sizeof(bf16));
    float* part  = (float*)carve((size_t)STATS_NB * 129 * 4);
    float* ss    = (float*)carve(128 * 4);
    if (off > ws_size) return;   // workspace too small; cannot proceed safely

    const float* vf    = (const float*)d_in[0];
    const int*   coors = (const int*)d_in[n_in - 2];
    const int    N     = in_sizes[0] / 16;

    // ---- helper lambdas ----
    auto conv = [&](int li, const bf16* xin, bf16* yout,
                    int Di,int Hi,int Wi, int Do,int Ho,int Wo,
                    int sd,int sh,int sw, int pd,int ph,int pw) {
        const CSpec& s = SP[li];
        const float* w = (const float*)d_in[1 + 3*li];
        int TAP = s.KD * s.KH * s.KW;
        int tot = TAP * s.Cin * s.Cout;
        bf16* wT = wbuf + (size_t)li * WSLOT;
        k_wconvert<<<(tot + 255)/256, 256, 0, stream>>>(w, TAP, s.Cin, s.Cout, wT);
        dim3 grid((Wo + 15)/16, Ho, Do);
        if (li == 0)
            k_conv_wmma<16,32,3,3,3><<<grid, 32, 0, stream>>>(
                xin, wT, yout, Di,Hi,Wi, Do,Ho,Wo, sd,sh,sw, pd,ph,pw);
        else if (li >= 1 && li <= 4)
            k_conv_wmma<32,32,3,3,3><<<grid, 32, 0, stream>>>(
                xin, wT, yout, Di,Hi,Wi, Do,Ho,Wo, sd,sh,sw, pd,ph,pw);
        else if (li == 5)
            k_conv_wmma<32,64,3,3,3><<<grid, 32, 0, stream>>>(
                xin, wT, yout, Di,Hi,Wi, Do,Ho,Wo, sd,sh,sw, pd,ph,pw);
        else if (li == 15)
            k_conv_wmma<64,64,3,1,1><<<grid, 32, 0, stream>>>(
                xin, wT, yout, Di,Hi,Wi, Do,Ho,Wo, sd,sh,sw, pd,ph,pw);
        else
            k_conv_wmma<64,64,3,3,3><<<grid, 32, 0, stream>>>(
                xin, wT, yout, Di,Hi,Wi, Do,Ho,Wo, sd,sh,sw, pd,ph,pw);
    };
    auto bn = [&](int li, const bf16* y, const float* mask, long nvox, int C,
                  const bf16* res, int relu, bf16* z) {
        int cshift = (C == 64) ? 6 : 5;
        k_stats_partial<<<STATS_NB, 256, 0, stream>>>(y, mask, nvox, cshift, part);
        const float* g = (const float*)d_in[1 + 3*li + 1];
        const float* b = (const float*)d_in[1 + 3*li + 2];
        k_stats_finalize<<<1, C, 0, stream>>>(part, STATS_NB, C, g, b, ss, 1e-3f);
        long tot = nvox * C;
        k_bn_apply<<<(unsigned)((tot + 255)/256), 256, 0, stream>>>(
            y, mask, ss, cshift, nvox, res, relu, z);
    };

    // ---- stage 0: scatter ----
    hipMemsetAsync(P[0], 0, (size_t)V0 * 16 * sizeof(bf16), stream);
    hipMemsetAsync(mask0, 0, (size_t)V0 * 4, stream);
    k_scatter<<<(N + 255)/256, 256, 0, stream>>>(vf, coors, N, H0, W0, 16, P[0], mask0);

    // ---- res0 (21,400,352), 32ch ----
    conv(0, P[0], P[1], D0,H0,W0, D0,H0,W0, 1,1,1, 1,1,1);
    bn(0, P[1], mask0, V0, 32, nullptr, 1, P[2]);
    conv(1, P[2], P[1], D0,H0,W0, D0,H0,W0, 1,1,1, 1,1,1);
    bn(1, P[1], mask0, V0, 32, nullptr, 1, P[3]);
    conv(2, P[3], P[1], D0,H0,W0, D0,H0,W0, 1,1,1, 1,1,1);
    bn(2, P[1], mask0, V0, 32, P[2],   1, P[0]);   // + residual (block r0a)
    conv(3, P[0], P[1], D0,H0,W0, D0,H0,W0, 1,1,1, 1,1,1);
    bn(3, P[1], mask0, V0, 32, nullptr, 1, P[2]);
    conv(4, P[2], P[1], D0,H0,W0, D0,H0,W0, 1,1,1, 1,1,1);
    bn(4, P[1], mask0, V0, 32, P[0],   1, P[3]);   // + residual (block r0b)

    // ---- down1 -> res1 (11,200,176), 64ch ----
    conv(5, P[3], P[1], D0,H0,W0, D1,H1,W1, 2,2,2, 1,1,1);
    k_mask_down<<<(unsigned)((V1 + 255)/256), 256, 0, stream>>>(
        mask0, mask1, D0,H0,W0, D1,H1,W1, 3,3,3, 2,2,2, 1,1,1);
    bn(5, P[1], mask1, V1, 64, nullptr, 1, P[0]);

    conv(6, P[0], P[1], D1,H1,W1, D1,H1,W1, 1,1,1, 1,1,1);
    bn(6, P[1], mask1, V1, 64, nullptr, 1, P[2]);
    conv(7, P[2], P[1], D1,H1,W1, D1,H1,W1, 1,1,1, 1,1,1);
    bn(7, P[1], mask1, V1, 64, P[0],   1, P[3]);   // block r1a
    conv(8, P[3], P[1], D1,H1,W1, D1,H1,W1, 1,1,1, 1,1,1);
    bn(8, P[1], mask1, V1, 64, nullptr, 1, P[2]);
    conv(9, P[2], P[1], D1,H1,W1, D1,H1,W1, 1,1,1, 1,1,1);
    bn(9, P[1], mask1, V1, 64, P[3],   1, P[0]);   // block r1b

    // ---- down2 -> res2 (5,100,88), 64ch ----
    conv(10, P[0], P[1], D1,H1,W1, D2,H2,W2, 2,2,2, 0,1,1);
    k_mask_down<<<(unsigned)((V2 + 255)/256), 256, 0, stream>>>(
        mask1, mask2, D1,H1,W1, D2,H2,W2, 3,3,3, 2,2,2, 0,1,1);
    bn(10, P[1], mask2, V2, 64, nullptr, 1, P[2]);

    conv(11, P[2], P[1], D2,H2,W2, D2,H2,W2, 1,1,1, 1,1,1);
    bn(11, P[1], mask2, V2, 64, nullptr, 1, P[3]);
    conv(12, P[3], P[1], D2,H2,W2, D2,H2,W2, 1,1,1, 1,1,1);
    bn(12, P[1], mask2, V2, 64, P[2],   1, P[0]);  // block r2a
    conv(13, P[0], P[1], D2,H2,W2, D2,H2,W2, 1,1,1, 1,1,1);
    bn(13, P[1], mask2, V2, 64, nullptr, 1, P[2]);
    conv(14, P[2], P[1], D2,H2,W2, D2,H2,W2, 1,1,1, 1,1,1);
    bn(14, P[1], mask2, V2, 64, P[0],   1, P[3]);  // block r2b

    // ---- down3 -> res3 (2,100,88), 64ch; final BN+relu+transpose to f32 out ----
    conv(15, P[3], P[1], D2,H2,W2, D3,H3,W3, 2,1,1, 0,0,0);
    k_mask_down<<<(unsigned)((V3 + 255)/256), 256, 0, stream>>>(
        mask2, mask3, D2,H2,W2, D3,H3,W3, 3,1,1, 2,1,1, 0,0,0);
    k_stats_partial<<<STATS_NB, 256, 0, stream>>>(P[1], mask3, V3, 6, part);
    k_stats_finalize<<<1, 64, 0, stream>>>(part, STATS_NB, 64,
        (const float*)d_in[1 + 3*15 + 1], (const float*)d_in[1 + 3*15 + 2], ss, 1e-3f);
    int totf = (int)(V3 * 64);
    k_bn_final<<<(totf + 255)/256, 256, 0, stream>>>(P[1], mask3, ss, 6, D3, H3, W3, (float*)d_out);
}
